// VectorQuantizer_38001870635819
// MI455X (gfx1250) — matrix operand order repaired and checked by reading, not compiled, not run on hardware
//
#include <hip/hip_runtime.h>

#define EDIM   256
#define NE     8192
#define NROWS  16384                 // 16*1024
#define NTILES (NE / 16)             // 512
#define KTILES (EDIM / 32)           // 8 k-tiles of K=32
#define ZQ_ELEMS (NROWS * EDIM)      // 4194304
#define BETA   0.25f
#define GATHER_BLOCKS (NROWS / 8)    // 2048
#define EPAD   (EDIM + 8)            // LDS row pad: conflict-spread, 16B aligned

#if defined(__has_builtin)
#  if __has_builtin(__builtin_amdgcn_global_load_async_to_lds_b128) && \
      __has_builtin(__builtin_amdgcn_s_wait_asynccnt)
#    define VQ_ASYNC 1
#  endif
#endif
#ifndef VQ_ASYNC
#  define VQ_ASYNC 0
#endif

#define AS1 __attribute__((address_space(1)))
#define AS3 __attribute__((address_space(3)))

typedef int    v4i  __attribute__((vector_size(16)));
typedef float  v2f  __attribute__((ext_vector_type(2)));
typedef float  v4f  __attribute__((ext_vector_type(4)));
typedef float  v8f  __attribute__((ext_vector_type(8)));
typedef __bf16 v4bf __attribute__((ext_vector_type(4)));
typedef __bf16 v8bf __attribute__((ext_vector_type(8)));
typedef __bf16 v16bf __attribute__((ext_vector_type(16)));

// Two 16B LDS loads -> one 32B B-fragment (rows are only 16B aligned).
static __device__ __forceinline__ v16bf lds_frag(const __bf16* p) {
    v8bf a = *(const v8bf*)p;
    v8bf b = *(const v8bf*)(p + 8);
    return __builtin_shufflevector(a, b, 0,1,2,3,4,5,6,7,8,9,10,11,12,13,14,15);
}

// ---------------------------------------------------------------------------
// Kernel 0: one-shot split of emb into bf16 hi/lo planes (hi=bf16(x),
// lo=bf16(x-hi)).  8 floats per thread; 1024 blocks x 256 threads.
// ---------------------------------------------------------------------------
__global__ __launch_bounds__(256) void vq_cvt(const float* __restrict__ emb,
                                              __bf16* __restrict__ ehi,
                                              __bf16* __restrict__ elo) {
    size_t i = ((size_t)blockIdx.x * 256 + threadIdx.x) * 8;
    v4f a = *(const v4f*)(emb + i);
    v4f b = *(const v4f*)(emb + i + 4);
    v8bf h, l;
#pragma unroll
    for (int e = 0; e < 8; ++e) {
        float x = (e < 4) ? a[e] : b[e - 4];
        __bf16 hx = (__bf16)x;
        h[e] = hx;
        l[e] = (__bf16)(x - (float)hx);
    }
    *(v8bf*)(ehi + i) = h;
    *(v8bf*)(elo + i) = l;
}

// ---------------------------------------------------------------------------
// Kernel 1: enorm[k] = sum_d emb[k][d]^2   (one wave per codebook row)
// ---------------------------------------------------------------------------
__global__ __launch_bounds__(256) void vq_enorm(const float* __restrict__ emb,
                                                float* __restrict__ enorm) {
    int wave = threadIdx.x >> 5;
    int lane = threadIdx.x & 31;
    int row  = blockIdx.x * 8 + wave;
    const float* p = emb + (size_t)row * EDIM;
    v4f a = *(const v4f*)(p + lane * 4);
    v4f b = *(const v4f*)(p + 128 + lane * 4);
    float s = a.x*a.x + a.y*a.y + a.z*a.z + a.w*a.w
            + b.x*b.x + b.y*b.y + b.z*b.z + b.w*b.w;
#pragma unroll
    for (int off = 16; off > 0; off >>= 1) s += __shfl_xor(s, off, 32);
    if (lane == 0) enorm[row] = s;
}

#if VQ_ASYNC
// Issue async global->LDS copies of one 16-code bf16 tile (hi+lo planes).
// Tile rows are contiguous in the planes: 512 chunks of 16B per plane.
static __device__ __forceinline__ void issue_tile_async(
        const __bf16* __restrict__ gh, const __bf16* __restrict__ gl,
        __bf16 (*bh)[EPAD], __bf16 (*bl)[EPAD], int tid) {
#pragma unroll
    for (int j = 0; j < 4; ++j) {
        int f = j * 128 + tid;        // 16B chunk id, 0..511
        int r = f >> 5, c = f & 31;
        __builtin_amdgcn_global_load_async_to_lds_b128(
            (AS1 v4i*)(gh + (size_t)f * 8), (AS3 v4i*)&bh[r][c * 8], 0, 0);
        __builtin_amdgcn_global_load_async_to_lds_b128(
            (AS1 v4i*)(gl + (size_t)f * 8), (AS3 v4i*)&bl[r][c * 8], 0, 0);
    }
}
#endif

// ---------------------------------------------------------------------------
// Kernel 2: argmin over codes using split-precision V_WMMA_F32_16X16X32_BF16.
// score(m,n) = enorm[n] - 2*z[m]·e[n];  dot = hi*hi + hi*lo + lo*hi in f32
// (dropped lo*lo ~2^-16 rel, negligible vs ~1e-4 inter-code separation).
// Block = 128 threads (4 waves); wave w owns rows [m0, m0+16).
// Double-buffered async global->LDS staging overlaps copy with WMMA.
// ---------------------------------------------------------------------------
__global__ __launch_bounds__(128) void vq_argmin(const float* __restrict__ z,
                                                 const __bf16* __restrict__ ehi,
                                                 const __bf16* __restrict__ elo,
                                                 const float* __restrict__ enorm,
                                                 int* __restrict__ idx_out) {
#if VQ_ASYNC
    __shared__ __bf16 sEhi[2][16][EPAD];
    __shared__ __bf16 sElo[2][16][EPAD];
#else
    __shared__ __bf16 sEhi[1][16][EPAD];
    __shared__ __bf16 sElo[1][16][EPAD];
#endif

    const int tid  = threadIdx.x;
    const int wave = tid >> 5;
    const int lane = tid & 31;
    const int col  = lane & 15;          // N within tile
    const int kh   = lane >> 4;          // K-half selector (16-bit A/B layout)
    const int m0   = blockIdx.x * 64 + wave * 16;

    // Build A fragments (-2*z, hi/lo split) in contiguous k order (same as B):
    // lane<16 holds K = {0..15} of each k-tile; lane>=16 holds {16..31}.
    v16bf a_hi[KTILES], a_lo[KTILES];
    const float* zrow = z + (size_t)(m0 + col) * EDIM;
#pragma unroll
    for (int kt = 0; kt < KTILES; ++kt) {
#pragma unroll
        for (int g = 0; g < 2; ++g) {
            int base = kt * 32 + 16 * kh + g * 8;
#pragma unroll
            for (int j = 0; j < 4; ++j) {
                v2f t = *(const v2f*)(zrow + base + 2 * j);
                float x0 = -2.0f * t.x, x1 = -2.0f * t.y;
                __bf16 h0 = (__bf16)x0, h1 = (__bf16)x1;
                a_hi[kt][g * 8 + 2 * j]     = h0;
                a_hi[kt][g * 8 + 2 * j + 1] = h1;
                a_lo[kt][g * 8 + 2 * j]     = (__bf16)(x0 - (float)h0);
                a_lo[kt][g * 8 + 2 * j + 1] = (__bf16)(x1 - (float)h1);
            }
        }
    }

    float bestS[8];
    int   bestI[8];
#pragma unroll
    for (int r = 0; r < 8; ++r) { bestS[r] = __builtin_inff(); bestI[r] = 0; }

#if VQ_ASYNC
    issue_tile_async(ehi, elo, sEhi[0], sElo[0], tid);   // prologue: tile 0
    __builtin_amdgcn_s_wait_asynccnt(0);
    __syncthreads();
#endif

    for (int t = 0; t < NTILES; ++t) {
        const int n0 = t * 16;
#if VQ_ASYNC
        const int buf = t & 1;
        if (t + 1 < NTILES)              // overlap next tile's DMA with compute
            issue_tile_async(ehi + (size_t)(t + 1) * 16 * EDIM,
                             elo + (size_t)(t + 1) * 16 * EDIM,
                             sEhi[buf ^ 1], sElo[buf ^ 1], tid);
#else
        const int buf = 0;
        __syncthreads();                 // previous tile's LDS reads done
#pragma unroll
        for (int j = 0; j < 4; ++j) {
            int f = j * 128 + tid;       // 16B chunk id
            int r = f >> 5, c = f & 31;
            v8bf h = *(const v8bf*)(ehi + (size_t)n0 * EDIM + (size_t)f * 8);
            v8bf l = *(const v8bf*)(elo + (size_t)n0 * EDIM + (size_t)f * 8);
            *(v8bf*)&sEhi[0][r][c * 8] = h;
            *(v8bf*)&sElo[0][r][c * 8] = l;
        }
        __syncthreads();
        if (t + 1 < NTILES)
            __builtin_prefetch(ehi + (size_t)(n0 + 16) * EDIM + tid * 32, 0, 0);
#endif

        float en = enorm[n0 + col];
        // Two zero-seeded accumulators (first WMMA uses inline C=0) to halve
        // the dependent chain; enorm added at the compare stage.
        v8f c0, c1;
#pragma unroll
        for (int r = 0; r < 8; ++r) { c0[r] = 0.0f; c1[r] = 0.0f; }

#pragma unroll
        for (int kt = 0; kt < KTILES; ++kt) {
            const __bf16* ph = &sEhi[buf][col][kt * 32 + 16 * kh];
            const __bf16* pl = &sElo[buf][col][kt * 32 + 16 * kh];
            v16bf bh = lds_frag(ph);
            v16bf bl = lds_frag(pl);
            v8f& cc = (kt & 1) ? c1 : c0;
            cc = __builtin_amdgcn_wmma_f32_16x16x32_bf16(
                     false, a_hi[kt], false, bh, (short)0, cc, false, false);
            cc = __builtin_amdgcn_wmma_f32_16x16x32_bf16(
                     false, a_hi[kt], false, bl, (short)0, cc, false, false);
            cc = __builtin_amdgcn_wmma_f32_16x16x32_bf16(
                     false, a_lo[kt], false, bh, (short)0, cc, false, false);
        }

        // Per-lane running argmin (strict < keeps first occurrence).
#pragma unroll
        for (int r = 0; r < 8; ++r) {
            float s = en + c0[r] + c1[r];
            if (s < bestS[r]) { bestS[r] = s; bestI[r] = n0 + col; }
        }

#if VQ_ASYNC
        __builtin_amdgcn_s_wait_asynccnt(0);   // next tile landed
        __syncthreads();                       // everyone done reading buf
#endif
    }

    // Cross-lane reduction: row (r + 8*kh) lives in 16 lanes of one half-wave.
#pragma unroll
    for (int r = 0; r < 8; ++r) {
        float s = bestS[r];
        int   i = bestI[r];
#pragma unroll
        for (int off = 8; off > 0; off >>= 1) {
            float s2 = __shfl_xor(s, off, 32);
            int   i2 = __shfl_xor(i, off, 32);
            if (s2 < s || (s2 == s && i2 < i)) { s = s2; i = i2; }
        }
        if (col == 0) idx_out[m0 + r + 8 * kh] = i;
    }
}

// ---------------------------------------------------------------------------
// Kernel 3: gather z_q = emb[idx], write idx as float, per-block partial SSE.
// One wave per row; deterministic in-block reduction (no float atomics).
// ---------------------------------------------------------------------------
__global__ __launch_bounds__(256) void vq_gather(const float* __restrict__ z,
                                                 const float* __restrict__ emb,
                                                 const int* __restrict__ idx,
                                                 float* __restrict__ out,
                                                 float* __restrict__ partial) {
    __shared__ float ssum[8];
    int wave = threadIdx.x >> 5;
    int lane = threadIdx.x & 31;
    int row  = blockIdx.x * 8 + wave;
    int code = idx[row];

    const float* ep = emb + (size_t)code * EDIM;
    const float* zp = z   + (size_t)row  * EDIM;
    float*       op = out + (size_t)row  * EDIM;

    float acc = 0.0f;
#pragma unroll
    for (int j = 0; j < 2; ++j) {
        int o  = j * 128 + lane * 4;
        v4f e  = *(const v4f*)(ep + o);
        v4f zz = *(const v4f*)(zp + o);
        *(v4f*)(op + o) = e;                 // straight-through fwd = emb[idx]
        v4f d = e - zz;
        acc += d.x*d.x + d.y*d.y + d.z*d.z + d.w*d.w;
    }
#pragma unroll
    for (int off = 16; off > 0; off >>= 1) acc += __shfl_xor(acc, off, 32);
    if (lane == 0) {
        ssum[wave] = acc;
        out[ZQ_ELEMS + 1 + row] = (float)code;   // idx section (as f32)
    }
    __syncthreads();
    if (threadIdx.x == 0) {
        float s = 0.0f;
#pragma unroll
        for (int w = 0; w < 8; ++w) s += ssum[w];
        partial[blockIdx.x] = s;
    }
}

// ---------------------------------------------------------------------------
// Kernel 4: deterministic tree-sum of partials; loss = (1+beta)*SSE/(N*D)
// ---------------------------------------------------------------------------
__global__ __launch_bounds__(256) void vq_loss(const float* __restrict__ partial,
                                               float* __restrict__ out) {
    __shared__ float red[256];
    float s = 0.0f;
    for (int i = threadIdx.x; i < GATHER_BLOCKS; i += 256) s += partial[i];
    red[threadIdx.x] = s;
    __syncthreads();
#pragma unroll
    for (int off = 128; off > 0; off >>= 1) {
        if (threadIdx.x < off) red[threadIdx.x] += red[threadIdx.x + off];
        __syncthreads();
    }
    if (threadIdx.x == 0)
        out[ZQ_ELEMS] = (1.0f + BETA) * red[0] / (float)((size_t)NROWS * EDIM);
}

// ---------------------------------------------------------------------------
extern "C" void kernel_launch(void* const* d_in, const int* in_sizes, int n_in,
                              void* d_out, int out_size, void* d_ws, size_t ws_size,
                              hipStream_t stream) {
    const float* z   = (const float*)d_in[0];   // [16,1024,256] f32
    const float* emb = (const float*)d_in[1];   // [8192,256]   f32
    float* out = (float*)d_out;                 // [zq | loss | idx]

    // Workspace layout (16B aligned sections):
    char*  ws      = (char*)d_ws;
    float* enorm   = (float*)ws;                                    // 32 KB
    int*   idx     = (int*)(ws + NE * sizeof(float));               // 64 KB
    float* partial = (float*)(ws + NE * sizeof(float)
                                 + NROWS * sizeof(int));            //  8 KB
    __bf16* ehi    = (__bf16*)(ws + NE * sizeof(float)
                                  + NROWS * sizeof(int)
                                  + GATHER_BLOCKS * sizeof(float)); //  4 MB
    __bf16* elo    = ehi + (size_t)NE * EDIM;                       //  4 MB

    vq_cvt   <<<(NE * EDIM) / 2048, 256, 0, stream>>>(emb, ehi, elo);
    vq_enorm <<<NE / 8,            256, 0, stream>>>(emb, enorm);
    vq_argmin<<<NROWS / 64,        128, 0, stream>>>(z, ehi, elo, enorm, idx);
    vq_gather<<<GATHER_BLOCKS,     256, 0, stream>>>(z, emb, idx, out, partial);
    vq_loss  <<<1,                 256, 0, stream>>>(partial, out);
}
